// DilatedAttention3D_6373731467413
// MI455X (gfx1250) — matrix operand order repaired and checked by reading, not compiled, not run on hardware
//
#include <hip/hip_runtime.h>
#include <math.h>

typedef __bf16 bf16_t;
typedef __bf16 v16bf __attribute__((ext_vector_type(16)));
typedef float  v8f   __attribute__((ext_vector_type(8)));
typedef unsigned int uint4v __attribute__((ext_vector_type(4)));

union FragU { v16bf v; uint4v u[2]; };

// A-matrix fragment (16x32 bf16, MxK). Caller passes per-lane pointer at
// [row(lane&15)][k0]. Per ISA table: lanes 0-15 hold K {0..7,16..23},
// lanes 16-31 hold K {8..15,24..31}.
__device__ __forceinline__ v16bf frag_a(const bf16_t* p, int lane) {
  int ko = (lane >> 4) << 3;
  FragU f;
  f.u[0] = *(const uint4v*)(p + ko);
  f.u[1] = *(const uint4v*)(p + ko + 16);
  return f.v;
}
// B-matrix fragment (32x16 bf16, KxN). Caller passes per-lane pointer at
// [col(lane&15)][k0] of the K-contiguous source. Lanes 0-15: K {0..7,16..23},
// lanes 16-31: K {8..15,24..31} (same order as A).
__device__ __forceinline__ v16bf frag_b(const bf16_t* p, int lane) {
  int ko = (lane >> 4) << 3;
  FragU f;
  f.u[0] = *(const uint4v*)(p + ko);
  f.u[1] = *(const uint4v*)(p + ko + 16);
  return f.v;
}

__device__ __forceinline__ v8f wmma_bf(v16bf a, v16bf b, v8f c) {
  return __builtin_amdgcn_wmma_f32_16x16x32_bf16(false, a, false, b, (short)0, c,
                                                 false, false);
}

// ---------------- GEMM: out[M,N] = A[M,K](bf16) * W[N,K]^T(bf16) + bias ------
template <int OUT_BF16>
__global__ __launch_bounds__(256) void gemm_kernel(
    const bf16_t* __restrict__ A, const bf16_t* __restrict__ W,
    const float* __restrict__ bias, void* __restrict__ outp,
    int M, int N, int K) {
  const int lane = threadIdx.x & 31;
  const int wave = threadIdx.x >> 5;
  const int nblk = N >> 7;
  const int bm = blockIdx.x / nblk;
  const int bn = blockIdx.x % nblk;
  const int m0 = bm * 128 + (wave >> 1) * 32;
  const int n0 = bn * 128 + (wave & 1) * 64;
  const int colLane = lane & 15;
  const int halfSel = lane >> 4;

  v8f acc[2][4];
#pragma unroll
  for (int i = 0; i < 2; ++i)
#pragma unroll
    for (int j = 0; j < 4; ++j)
#pragma unroll
      for (int v = 0; v < 8; ++v) acc[i][j][v] = 0.0f;

  const bf16_t* arow0 = A + (size_t)(m0 + colLane) * K;
  const bf16_t* arow1 = A + (size_t)(m0 + 16 + colLane) * K;
  for (int k = 0; k < K; k += 32) {
    v16bf a0 = frag_a(arow0 + k, lane);
    v16bf a1 = frag_a(arow1 + k, lane);
#pragma unroll
    for (int j = 0; j < 4; ++j) {
      v16bf bb = frag_b(W + (size_t)(n0 + j * 16 + colLane) * K + k, lane);
      acc[0][j] = wmma_bf(a0, bb, acc[0][j]);
      acc[1][j] = wmma_bf(a1, bb, acc[1][j]);
    }
  }
#pragma unroll
  for (int i = 0; i < 2; ++i)
#pragma unroll
    for (int j = 0; j < 4; ++j) {
      int n = n0 + j * 16 + colLane;
      float bv = bias[n];
#pragma unroll
      for (int v = 0; v < 8; ++v) {
        int m = m0 + i * 16 + halfSel * 8 + v;
        float val = acc[i][j][v] + bv;
        if (OUT_BF16)
          ((bf16_t*)outp)[(size_t)m * N + n] = (bf16_t)val;
        else
          ((float*)outp)[(size_t)m * N + n] = val;
      }
    }
}

// ---------------- Dilated window attention, one scale ------------------------
// One wave handles a 16-query strip of one (window, head) problem of N=512
// tokens, head_dim=64. Exact softmax (full score row in registers).
__global__ __launch_bounds__(256) void attn_kernel(
    const bf16_t* __restrict__ Qb, const bf16_t* __restrict__ Kb,
    const bf16_t* __restrict__ Vb, const float* __restrict__ mask,
    bf16_t* __restrict__ Obuf, float* __restrict__ Lse,
    int ws, int r, int nwd) {
  __shared__ __align__(16) bf16_t Vt[64 * 32];      // [ch][key] transposed V blk
  __shared__ __align__(16) bf16_t Pst[8][16 * 32];  // per-wave P staging

  const int tid = threadIdx.x;
  const int lane = tid & 31;
  const int wave = tid >> 5;
  const int colLane = lane & 15;
  const int halfSel = lane >> 4;

  const int blk = blockIdx.x;
  const int qc = blk & 3;       // 4 query-chunks of 128 per problem
  const int wh = blk >> 2;
  const int h = wh & 7;
  const int win = wh >> 3;
  const int wx = win % nwd;
  const int tmpw = win / nwd;
  const int wy = tmpw % nwd;
  const int wz = tmpw / nwd;
  int od = 0, oh2 = 0, ow = 0;
  if (r == 2) { od = h >> 2; oh2 = (h >> 1) & 1; ow = h & 1; }
  const int bz = wz * ws + od, by = wy * ws + oh2, bx2 = wx * ws + ow;

  auto tok = [&](int n) -> int {
    int zs = n >> 6, ys = (n >> 3) & 7, xs = n & 7;
    return (bz + zs * r) * 1024 + (by + ys * r) * 32 + (bx2 + xs * r);
  };

  const int qbase = qc * 128 + wave * 16;
  const int tq = tok(qbase + colLane);
  const bf16_t* qrow = Qb + (size_t)tq * 512 + h * 64;
  v16bf qa0 = frag_a(qrow, lane);
  v16bf qa1 = frag_a(qrow + 32, lane);

  // ---- scores: S[nt] tile covers keys nt*16..nt*16+15 ----
  v8f S[32];
#pragma unroll
  for (int nt = 0; nt < 32; ++nt) {
    int tk = tok(nt * 16 + colLane);
    const bf16_t* krow = Kb + (size_t)tk * 512 + h * 64;
    v8f s;
#pragma unroll
    for (int v = 0; v < 8; ++v) s[v] = 0.0f;
    s = wmma_bf(qa0, frag_b(krow, lane), s);
    s = wmma_bf(qa1, frag_b(krow + 32, lane), s);
    float biasv = (mask[tk] == 0.0f) ? -1e9f : 0.0f;
#pragma unroll
    for (int v = 0; v < 8; ++v) S[nt][v] = s[v] * 0.125f + biasv;
  }

  // ---- row max (across 32 tiles, then across the 16 lanes of the half) ----
  v8f m8 = S[0];
#pragma unroll
  for (int nt = 1; nt < 32; ++nt)
#pragma unroll
    for (int v = 0; v < 8; ++v) m8[v] = fmaxf(m8[v], S[nt][v]);
#pragma unroll
  for (int off = 1; off < 16; off <<= 1)
#pragma unroll
    for (int v = 0; v < 8; ++v) m8[v] = fmaxf(m8[v], __shfl_xor(m8[v], off, 16));

  // ---- exp + row sum ----
  v8f sum8;
#pragma unroll
  for (int v = 0; v < 8; ++v) sum8[v] = 0.0f;
#pragma unroll
  for (int nt = 0; nt < 32; ++nt)
#pragma unroll
    for (int v = 0; v < 8; ++v) {
      float e = expf(S[nt][v] - m8[v]);
      S[nt][v] = e;
      sum8[v] += e;
    }
#pragma unroll
  for (int off = 1; off < 16; off <<= 1)
#pragma unroll
    for (int v = 0; v < 8; ++v) sum8[v] += __shfl_xor(sum8[v], off, 16);

  if (colLane == 0) {
#pragma unroll
    for (int v = 0; v < 8; ++v) {
      int qn = qbase + halfSel * 8 + v;
      Lse[h * 32768 + tok(qn)] = m8[v] + logf(sum8[v]);
    }
  }

  // ---- O = P * V over 16 key-blocks of 32 ----
  v8f O[4];
#pragma unroll
  for (int ct = 0; ct < 4; ++ct)
#pragma unroll
    for (int v = 0; v < 8; ++v) O[ct][v] = 0.0f;

  bf16_t* ps = &Pst[wave][0];
#pragma unroll
  for (int kb = 0; kb < 16; ++kb) {
    __syncthreads();
    {  // cooperative transposed V block load: Vt[ch][key]
      int key = tid >> 3;
      int chb = (tid & 7) * 8;
      const bf16_t* vp = Vb + (size_t)tok(kb * 32 + key) * 512 + h * 64 + chb;
      uint4v vv = *(const uint4v*)vp;
      const bf16_t* vsrc = (const bf16_t*)&vv;
#pragma unroll
      for (int e = 0; e < 8; ++e) Vt[(chb + e) * 32 + key] = vsrc[e];
    }
    __syncthreads();
    // stage P (C-layout -> row-major 16x32 bf16) for this wave
#pragma unroll
    for (int t2 = 0; t2 < 2; ++t2)
#pragma unroll
      for (int v = 0; v < 8; ++v)
        ps[(halfSel * 8 + v) * 32 + t2 * 16 + colLane] =
            (bf16_t)S[kb * 2 + t2][v];
    asm volatile("" ::: "memory");  // keep LDS store->load order
    v16bf pa = frag_a(ps + colLane * 32, lane);
#pragma unroll
    for (int ct = 0; ct < 4; ++ct) {
      v16bf vbf = frag_b(Vt + (ct * 16 + colLane) * 32, lane);
      O[ct] = wmma_bf(pa, vbf, O[ct]);
    }
  }

#pragma unroll
  for (int v = 0; v < 8; ++v) {
    int qn = qbase + halfSel * 8 + v;
    size_t base = ((size_t)h * 32768 + tok(qn)) * 64;
    float inv = 1.0f / sum8[v];
#pragma unroll
    for (int ct = 0; ct < 4; ++ct)
      Obuf[base + ct * 16 + colLane] = (bf16_t)(O[ct][v] * inv);
  }
}

// ---------------- LSE-weighted merge across the two scales -------------------
__global__ void merge_kernel(const bf16_t* __restrict__ o1,
                             const bf16_t* __restrict__ o2,
                             const float* __restrict__ l1,
                             const float* __restrict__ l2,
                             const float* __restrict__ mask,
                             bf16_t* __restrict__ y) {
  size_t i = (size_t)blockIdx.x * 256 + threadIdx.x;
  if (i >= (size_t)32768 * 512) return;
  int c = (int)(i & 511);
  int t = (int)(i >> 9);
  int h = c >> 6;
  int cc = c & 63;
  size_t li = (size_t)h * 32768 + t;
  float a = l1[li], b = l2[li];
  float m = fmaxf(a, b);
  float w1 = expf(a - m), w2 = expf(b - m);
  float z = w1 + w2;
  size_t oi = li * 64 + cc;
  float v1 = (float)o1[oi];
  float v2 = (float)o2[oi];
  y[i] = (bf16_t)(mask[t] * (w1 * v1 + w2 * v2) / z);
}

// ---------------- small utility kernels --------------------------------------
__global__ void f32_to_bf16(const float* __restrict__ s, bf16_t* __restrict__ d,
                            size_t n) {
  size_t i = (size_t)blockIdx.x * 256 + threadIdx.x;
  if (i < n) d[i] = (bf16_t)s[i];
}
__global__ void fill_f32(float* __restrict__ p, float v, size_t n) {
  size_t i = (size_t)blockIdx.x * 256 + threadIdx.x;
  if (i < n) p[i] = v;
}

extern "C" void kernel_launch(void* const* d_in, const int* in_sizes, int n_in,
                              void* d_out, int out_size, void* d_ws,
                              size_t ws_size, hipStream_t stream) {
  const float* x    = (const float*)d_in[0];
  const float* mask = (const float*)d_in[1];
  const float* Wq   = (const float*)d_in[2];
  const float* bq   = (const float*)d_in[3];
  const float* Wk   = (const float*)d_in[4];
  const float* bk   = (const float*)d_in[5];
  const float* Wv   = (const float*)d_in[6];
  const float* bv   = (const float*)d_in[7];
  const float* Wo   = (const float*)d_in[8];
  const float* bo   = (const float*)d_in[9];

  const size_t L = 32768, C = 512;
  char* ws = (char*)d_ws;
  size_t off = 0;
  auto alloc = [&](size_t bytes) -> void* {
    void* p = ws + off;
    off = (off + bytes + 255) & ~(size_t)255;
    return p;
  };
  bf16_t* xb  = (bf16_t*)alloc(L * C * 2);
  bf16_t* wqb = (bf16_t*)alloc(C * C * 2);
  bf16_t* wkb = (bf16_t*)alloc(C * C * 2);
  bf16_t* wvb = (bf16_t*)alloc(C * C * 2);
  bf16_t* wob = (bf16_t*)alloc(C * C * 2);
  bf16_t* qb  = (bf16_t*)alloc(L * C * 2);
  bf16_t* kb  = (bf16_t*)alloc(L * C * 2);
  bf16_t* vb2 = (bf16_t*)alloc(L * C * 2);
  bf16_t* o1  = (bf16_t*)alloc(L * C * 2);
  bf16_t* o2  = (bf16_t*)alloc(L * C * 2);
  float*  l1  = (float*)alloc(8 * L * 4);
  float*  l2  = (float*)alloc(8 * L * 4);
  bf16_t* yb  = (bf16_t*)alloc(L * C * 2);

  // bf16 conversions of activations and weights
  f32_to_bf16<<<(unsigned)((L * C + 255) / 256), 256, 0, stream>>>(x, xb, L * C);
  f32_to_bf16<<<(unsigned)((C * C + 255) / 256), 256, 0, stream>>>(Wq, wqb, C * C);
  f32_to_bf16<<<(unsigned)((C * C + 255) / 256), 256, 0, stream>>>(Wk, wkb, C * C);
  f32_to_bf16<<<(unsigned)((C * C + 255) / 256), 256, 0, stream>>>(Wv, wvb, C * C);
  f32_to_bf16<<<(unsigned)((C * C + 255) / 256), 256, 0, stream>>>(Wo, wob, C * C);
  // scale-2 LSE default = LSE_FILL for positions outside each head's grid
  fill_f32<<<(unsigned)((8 * L + 255) / 256), 256, 0, stream>>>(l2, -1e8f, 8 * L);

  const unsigned gemmGrid = (unsigned)((L / 128) * (C / 128));  // 1024
  gemm_kernel<1><<<gemmGrid, 256, 0, stream>>>(xb, wqb, bq, qb, (int)L, (int)C, (int)C);
  gemm_kernel<1><<<gemmGrid, 256, 0, stream>>>(xb, wkb, bk, kb, (int)L, (int)C, (int)C);
  gemm_kernel<1><<<gemmGrid, 256, 0, stream>>>(xb, wvb, bv, vb2, (int)L, (int)C, (int)C);

  // scale 1: window 8^3, dilation 1, 4^3 windows -> 64*8 problems * 4 chunks
  attn_kernel<<<64 * 8 * 4, 256, 0, stream>>>(qb, kb, vb2, mask, o1, l1, 8, 1, 4);
  // scale 2: window 16^3, dilation 2, 2^3 windows -> 8*8 problems * 4 chunks
  attn_kernel<<<8 * 8 * 4, 256, 0, stream>>>(qb, kb, vb2, mask, o2, l2, 16, 2, 2);

  merge_kernel<<<(unsigned)((L * C + 255) / 256), 256, 0, stream>>>(o1, o2, l1, l2,
                                                                    mask, yb);
  gemm_kernel<0><<<gemmGrid, 256, 0, stream>>>(yb, wob, bo, (float*)d_out, (int)L,
                                               (int)C, (int)C);
}
